// RWKV_88983132438845
// MI455X (gfx1250) — matrix operand order repaired and checked by reading, not compiled, not run on hardware
//
#include <hip/hip_runtime.h>

// ---------------------------------------------------------------------------
// RWKV-4 forward for gfx1250 (MI455X).
// GEMMs: v_wmma_f32_16x16x32_bf16 fed from LDS, staged with
// global_load_async_to_lds_b128 (ASYNCcnt) and double buffering.
// ---------------------------------------------------------------------------

#define LL 4
#define BB 4
#define TT 2048
#define CC 1024
#define FF 4096
#define MTOK (BB * TT)   // 8192 tokens

typedef unsigned short u16;
typedef __attribute__((ext_vector_type(4))) unsigned int u32x4;
typedef __attribute__((ext_vector_type(16))) __bf16 v16bf;
typedef __attribute__((ext_vector_type(8)))  float  v8f;

union Frag16 {           // one WMMA 16-bit operand: 16 bf16 = 32 bytes
    v16bf v;
    u32x4 q[2];
};

__device__ __forceinline__ u16 f2bf(float f) {
    unsigned int u = __float_as_uint(f);
    unsigned int r = u + 0x7FFFu + ((u >> 16) & 1u);   // round-to-nearest-even
    return (u16)(r >> 16);
}

__device__ __forceinline__ float sigmoidf_(float x) {
    return 1.0f / (1.0f + __expf(-x));
}

// ---------------------------------------------------------------------------
// LayerNorm: one 256-thread block per token, C = 1024 (4 elems/thread).
// ---------------------------------------------------------------------------
__global__ __launch_bounds__(256)
void ln_kernel(const float* __restrict__ x, const float* __restrict__ w,
               const float* __restrict__ b, float* __restrict__ out) {
    const int tok = blockIdx.x;
    const float* xp = x + (size_t)tok * CC;
    float* op = out + (size_t)tok * CC;

    float vals[4];
    float s1 = 0.f, s2 = 0.f;
#pragma unroll
    for (int i = 0; i < 4; ++i) {
        float v = xp[threadIdx.x + i * 256];
        vals[i] = v;
        s1 += v;
        s2 += v * v;
    }
#pragma unroll
    for (int off = 16; off > 0; off >>= 1) {
        s1 += __shfl_down(s1, off, 32);
        s2 += __shfl_down(s2, off, 32);
    }
    __shared__ float red1[8], red2[8];
    const int wave = threadIdx.x >> 5;
    if ((threadIdx.x & 31) == 0) { red1[wave] = s1; red2[wave] = s2; }
    __syncthreads();
    float t1 = 0.f, t2 = 0.f;
#pragma unroll
    for (int i = 0; i < 8; ++i) { t1 += red1[i]; t2 += red2[i]; }
    const float mean = t1 * (1.0f / CC);
    const float var  = t2 * (1.0f / CC) - mean * mean;
    const float rstd = rsqrtf(var + 1e-5f);
#pragma unroll
    for (int i = 0; i < 4; ++i) {
        int c = threadIdx.x + i * 256;
        op[c] = (vals[i] - mean) * rstd * w[c] + b[c];
    }
}

// ---------------------------------------------------------------------------
// Time-shift mixes -> bf16 GEMM operands.  Layout [B,T,C]; shift = prev token.
// ---------------------------------------------------------------------------
__global__ __launch_bounds__(256)
void mix3_kernel(const float* __restrict__ h,
                 const float* __restrict__ mk, const float* __restrict__ mv,
                 const float* __restrict__ mr,
                 u16* __restrict__ xk, u16* __restrict__ xv, u16* __restrict__ xr) {
    const int idx = blockIdx.x * 256 + threadIdx.x;   // over B*T*C
    const int c = idx & (CC - 1);
    const int t = (idx >> 10) & (TT - 1);
    const float hc = h[idx];
    const float hp = (t == 0) ? 0.f : h[idx - CC];
    const float ak = mk[c], av = mv[c], ar = mr[c];
    xk[idx] = f2bf(hc * ak + hp * (1.f - ak));
    xv[idx] = f2bf(hc * av + hp * (1.f - av));
    xr[idx] = f2bf(hc * ar + hp * (1.f - ar));
}

__global__ __launch_bounds__(256)
void mix2_kernel(const float* __restrict__ h,
                 const float* __restrict__ mk, const float* __restrict__ mr,
                 u16* __restrict__ xk, u16* __restrict__ xr) {
    const int idx = blockIdx.x * 256 + threadIdx.x;
    const int c = idx & (CC - 1);
    const int t = (idx >> 10) & (TT - 1);
    const float hc = h[idx];
    const float hp = (t == 0) ? 0.f : h[idx - CC];
    const float ak = mk[c], ar = mr[c];
    xk[idx] = f2bf(hc * ak + hp * (1.f - ak));
    xr[idx] = f2bf(hc * ar + hp * (1.f - ar));
}

// ---------------------------------------------------------------------------
// fp32 -> bf16 weight conversion
// ---------------------------------------------------------------------------
__global__ __launch_bounds__(256)
void conv_bf16_kernel(const float* __restrict__ src, u16* __restrict__ dst, int n) {
    int idx = blockIdx.x * 256 + threadIdx.x;
    if (idx < n) dst[idx] = f2bf(src[idx]);
}

// ---------------------------------------------------------------------------
// WKV recurrence: one thread per (b,c) channel, numerically-stable scan over T.
// Fuses rwkv = sigmoid(r) * wkv and emits bf16 for the Wo GEMM.
// ---------------------------------------------------------------------------
__global__ __launch_bounds__(256)
void wkv_kernel(const float* __restrict__ td, const float* __restrict__ tf,
                const float* __restrict__ k, const float* __restrict__ v,
                const float* __restrict__ sr, u16* __restrict__ rwkv) {
    const int gid = blockIdx.x * 256 + threadIdx.x;   // 0 .. B*C-1
    const int c = gid & (CC - 1);
    const int b = gid >> 10;
    const float w = -__expf(td[c]);
    const float u = tf[c];
    float aa = 0.f, bb = 0.f, pp = -1e38f;
    size_t base = (size_t)b * TT * CC + c;
    for (int t = 0; t < TT; ++t) {
        const size_t i = base + (size_t)t * CC;
        const float kk = k[i], vv = v[i];
        const float ww = u + kk;
        const float qq = fmaxf(pp, ww);
        const float e1 = __expf(pp - qq);
        const float e2 = __expf(ww - qq);
        const float out = (e1 * aa + e2 * vv) / (e1 * bb + e2);
        const float ww2 = pp + w;
        const float qq2 = fmaxf(ww2, kk);
        const float e1b = __expf(ww2 - qq2);
        const float e2b = __expf(kk - qq2);
        aa = e1b * aa + e2b * vv;
        bb = e1b * bb + e2b;
        pp = qq2;
        rwkv[i] = f2bf(sr[i] * out);
    }
}

// ---------------------------------------------------------------------------
// NT GEMM on WMMA with async-LDS staging:
//   out[M,N] = A[M,K] * W[N,K]^T   (bf16 in, fp32 accumulate)
// Block = 256 threads = 8 waves (2 M x 4 N).  Block tile 128x128, K-step 32.
// LDS: As[128][32] + Bs[128][32] bf16, double buffered (32 KB total).
// Staging: global_load_async_to_lds_b128 (4 per thread per K-step),
// synchronized with s_wait_asynccnt + workgroup barriers.
// Wave tile 64x32 = 4x2 WMMA 16x16x32 tiles.  All 6 fragments (4 A + 2 B) are
// loaded via ds_load_b128 BEFORE the WMMA chain so the 8 WMMAs issue
// back-to-back without per-tile s_wait_dscnt stalls.
//   A 16x32 : lane m=lane&15, kh=lane>>4 -> 16B at row*64 + kh*16 and +32
//   B 32x16 : lane n=lane&15, kh=lane>>4 -> 16B at row*64 + kh*16 and +32
//   D 16x16 : VGPR r -> row kh*8+r, col lane&15
// ---------------------------------------------------------------------------
enum { EPI_NONE = 0, EPI_SIGMOID = 1, EPI_RELUSQ_BF16 = 2, EPI_ADD = 3, EPI_MULADD = 4 };

#define LDS_BUF_BYTES 16384      // per double-buffer slot: A 8KB + B 8KB

template <int EPI>
__global__ __launch_bounds__(256)
void gemm_nt_kernel(const u16* __restrict__ A, const u16* __restrict__ W,
                    void* __restrict__ out, const float* __restrict__ aux,
                    int M, int N, int K) {
    __shared__ u16 smem[2 * LDS_BUF_BYTES / 2];   // 32 KB
    char* sm = (char*)smem;

    const int tid  = threadIdx.x;
    const int lane = tid & 31;
    const int wave = tid >> 5;
    const int waveM = wave & 1;        // 0..1
    const int waveN = wave >> 1;       // 0..3
    const int m0 = blockIdx.x * 128;
    const int n0 = blockIdx.y * 128;
    const int lm = lane & 15;
    const int kh = lane >> 4;

    v8f acc[4][2] = {};

    // --- async staging: copy A[128][32] and B[128][32] bf16 tiles to LDS ---
    auto issue = [&](int p, int k0) {
        const unsigned abase = (unsigned)(p * LDS_BUF_BYTES);
        const unsigned bbase = abase + 8192u;
#pragma unroll
        for (int s = 0; s < 2; ++s) {
            const int segi = tid * 2 + s;          // 0..511
            const int row  = segi >> 2;            // 0..127
            const int sg   = segi & 3;             // 16B segment within row
            const unsigned long long ga =
                (unsigned long long)(A + (size_t)(m0 + row) * K + k0 + sg * 8);
            const unsigned la = abase + (unsigned)(row * 64 + sg * 16);
            asm volatile("global_load_async_to_lds_b128 %0, %1, off"
                         :: "v"(la), "v"(ga) : "memory");
            const unsigned long long gb =
                (unsigned long long)(W + (size_t)(n0 + row) * K + k0 + sg * 8);
            const unsigned lb = bbase + (unsigned)(row * 64 + sg * 16);
            asm volatile("global_load_async_to_lds_b128 %0, %1, off"
                         :: "v"(lb), "v"(gb) : "memory");
        }
    };

    const int NK = K >> 5;
    issue(0, 0);                                   // prologue: buffer 0

    for (int ks = 0; ks < NK; ++ks) {
        const int p = ks & 1;
        if (ks + 1 < NK) {
            issue(p ^ 1, (ks + 1) * 32);
            asm volatile("s_wait_asynccnt 0x4" ::: "memory");   // oldest buf done
        } else {
            asm volatile("s_wait_asynccnt 0x0" ::: "memory");
        }
        __syncthreads();

        const char* Ab = sm + p * LDS_BUF_BYTES;
        const char* Bb = Ab + 8192;

        // Load ALL fragments first (12 ds_load_b128), then 8 WMMAs unbroken.
        Frag16 bfr[2], afr[4];
#pragma unroll
        for (int j = 0; j < 2; ++j) {
            const int row = waveN * 32 + j * 16 + lm;
            bfr[j].q[0] = *(const u32x4*)(Bb + row * 64 + kh * 16);
            bfr[j].q[1] = *(const u32x4*)(Bb + row * 64 + kh * 16 + 32);
        }
#pragma unroll
        for (int i = 0; i < 4; ++i) {
            const int row = waveM * 64 + i * 16 + lm;
            afr[i].q[0] = *(const u32x4*)(Ab + row * 64 + kh * 16);
            afr[i].q[1] = *(const u32x4*)(Ab + row * 64 + kh * 16 + 32);
        }
#pragma unroll
        for (int i = 0; i < 4; ++i) {
            acc[i][0] = __builtin_amdgcn_wmma_f32_16x16x32_bf16(
                false, afr[i].v, false, bfr[0].v, (short)0, acc[i][0], false, false);
            acc[i][1] = __builtin_amdgcn_wmma_f32_16x16x32_bf16(
                false, afr[i].v, false, bfr[1].v, (short)0, acc[i][1], false, false);
        }
        __syncthreads();   // protect LDS buffer reuse next iteration
    }

    // --- epilogue ---
    const int wm = m0 + waveM * 64;
    const int wn = n0 + waveN * 32;
#pragma unroll
    for (int i = 0; i < 4; ++i) {
#pragma unroll
        for (int j = 0; j < 2; ++j) {
#pragma unroll
            for (int r = 0; r < 8; ++r) {
                const int m = wm + i * 16 + kh * 8 + r;
                const int n = wn + j * 16 + lm;
                const size_t o = (size_t)m * N + n;
                const float val = acc[i][j][r];
                if constexpr (EPI == EPI_NONE) {
                    ((float*)out)[o] = val;
                } else if constexpr (EPI == EPI_SIGMOID) {
                    ((float*)out)[o] = sigmoidf_(val);
                } else if constexpr (EPI == EPI_RELUSQ_BF16) {
                    const float rl = fmaxf(val, 0.f);
                    ((u16*)out)[o] = f2bf(rl * rl);
                } else if constexpr (EPI == EPI_ADD) {
                    ((float*)out)[o] += val;
                } else {  // EPI_MULADD
                    ((float*)out)[o] += aux[o] * val;
                }
            }
        }
    }
}

// ---------------------------------------------------------------------------
// Host orchestration
// ---------------------------------------------------------------------------
extern "C" void kernel_launch(void* const* d_in, const int* in_sizes, int n_in,
                              void* d_out, int out_size, void* d_ws, size_t ws_size,
                              hipStream_t stream) {
    (void)in_sizes; (void)n_in; (void)out_size; (void)ws_size;

    const float* x_in   = (const float*)d_in[0];
    const float* ln0_w  = (const float*)d_in[1];
    const float* ln0_b  = (const float*)d_in[2];
    const float* ln1_w  = (const float*)d_in[3];
    const float* ln1_b  = (const float*)d_in[4];
    const float* ln2_w  = (const float*)d_in[5];
    const float* ln2_b  = (const float*)d_in[6];
    const float* tdec   = (const float*)d_in[7];
    const float* tfirst = (const float*)d_in[8];
    const float* tmk    = (const float*)d_in[9];
    const float* tmv    = (const float*)d_in[10];
    const float* tmr    = (const float*)d_in[11];
    const float* Wk     = (const float*)d_in[12];
    const float* Wv     = (const float*)d_in[13];
    const float* Wr     = (const float*)d_in[14];
    const float* Wo     = (const float*)d_in[15];
    const float* cmk    = (const float*)d_in[16];
    const float* cmr    = (const float*)d_in[17];
    const float* Wck    = (const float*)d_in[18];
    const float* Wcv    = (const float*)d_in[19];
    const float* Wcr    = (const float*)d_in[20];

    float* x = (float*)d_out;                 // residual stream lives in d_out
    char*  ws = (char*)d_ws;
    constexpr size_t MB = 1ull << 20;

    float* h      = (float*)(ws + 0 * MB);    // 32 MB  fp32 [B,T,C]
    u16*   xk     = (u16*)  (ws + 32 * MB);   // 16 MB  bf16 [B*T,C]
    u16*   xv     = (u16*)  (ws + 48 * MB);   // 16 MB
    u16*   xr     = (u16*)  (ws + 64 * MB);   // 16 MB
    float* kbuf   = (float*)(ws + 80 * MB);   // 32 MB  fp32 [B,T,C]
    float* vbuf   = (float*)(ws + 112 * MB);  // 32 MB
    float* srb    = (float*)(ws + 144 * MB);  // 32 MB  sigmoid(r) fp32
    u16*   rwkvb  = (u16*)  (ws + 176 * MB);  // 16 MB  bf16
    u16*   kf     = (u16*)  (ws + 192 * MB);  // 64 MB  bf16 [B*T,F]
    u16*   wb     = (u16*)  (ws + 256 * MB);  // 18 MB  per-layer bf16 weights

    const int nElem = MTOK * CC;              // 8,388,608
    const dim3 blk(256);
    const dim3 gLN(MTOK);
    const dim3 gMix(nElem / 256);
    const dim3 gWkv((BB * CC) / 256);
    const dim3 gGemmC(MTOK / 128, CC / 128);  // N = 1024
    const dim3 gGemmF(MTOK / 128, FF / 128);  // N = 4096

    auto conv = [&](const float* s, u16* d, int n) {
        conv_bf16_kernel<<<dim3((n + 255) / 256), blk, 0, stream>>>(s, d, n);
    };

    // ln0
    ln_kernel<<<gLN, blk, 0, stream>>>(x_in, ln0_w, ln0_b, x);

    for (int l = 0; l < LL; ++l) {
        const size_t lC = (size_t)l * CC;
        const size_t lCC = (size_t)l * CC * CC;
        const size_t lFC = (size_t)l * FF * CC;

        // ---------------- TimeMix ----------------
        ln_kernel<<<gLN, blk, 0, stream>>>(x, ln1_w + lC, ln1_b + lC, h);
        mix3_kernel<<<gMix, blk, 0, stream>>>(h, tmk + lC, tmv + lC, tmr + lC,
                                              xk, xv, xr);

        u16* wbk = wb;
        u16* wbv = wb + (size_t)CC * CC;
        u16* wbr = wb + 2 * (size_t)CC * CC;
        u16* wbo = wb + 3 * (size_t)CC * CC;
        conv(Wk + lCC, wbk, CC * CC);
        conv(Wv + lCC, wbv, CC * CC);
        conv(Wr + lCC, wbr, CC * CC);
        conv(Wo + lCC, wbo, CC * CC);

        gemm_nt_kernel<EPI_NONE><<<gGemmC, blk, 0, stream>>>(
            xk, wbk, kbuf, nullptr, MTOK, CC, CC);
        gemm_nt_kernel<EPI_NONE><<<gGemmC, blk, 0, stream>>>(
            xv, wbv, vbuf, nullptr, MTOK, CC, CC);
        gemm_nt_kernel<EPI_SIGMOID><<<gGemmC, blk, 0, stream>>>(
            xr, wbr, srb, nullptr, MTOK, CC, CC);

        wkv_kernel<<<gWkv, blk, 0, stream>>>(tdec + lC, tfirst + lC,
                                             kbuf, vbuf, srb, rwkvb);

        gemm_nt_kernel<EPI_ADD><<<gGemmC, blk, 0, stream>>>(
            rwkvb, wbo, x, nullptr, MTOK, CC, CC);

        // ---------------- ChannelMix ----------------
        ln_kernel<<<gLN, blk, 0, stream>>>(x, ln2_w + lC, ln2_b + lC, h);
        mix2_kernel<<<gMix, blk, 0, stream>>>(h, cmk + lC, cmr + lC, xk, xr);

        u16* wbck = wb;                       // [F, C]
        u16* wbcv = wb + (size_t)FF * CC;     // [C, F]
        u16* wbcr = wb + 2 * (size_t)FF * CC; // [C, C]
        conv(Wck + lFC, wbck, FF * CC);
        conv(Wcv + lFC, wbcv, CC * FF);
        conv(Wcr + lCC, wbcr, CC * CC);

        gemm_nt_kernel<EPI_RELUSQ_BF16><<<gGemmF, blk, 0, stream>>>(
            xk, wbck, kf, nullptr, MTOK, FF, CC);
        gemm_nt_kernel<EPI_SIGMOID><<<gGemmC, blk, 0, stream>>>(
            xr, wbcr, srb, nullptr, MTOK, CC, CC);
        gemm_nt_kernel<EPI_MULADD><<<gGemmC, blk, 0, stream>>>(
            kf, wbcv, x, srb, MTOK, CC, FF);
    }
}
